// GraphTransformer_9259949490194
// MI455X (gfx1250) — matrix operand order repaired and checked by reading, not compiled, not run on hardware
//
#include <hip/hip_runtime.h>
#include <hip/hip_bf16.h>
#include <math.h>

// Problem constants (L,B,S,E,H,F from the reference)
#define NL 2
#define NB 2
#define NS 128
#define NE 512
#define NH 8
#define ND 64
#define NF 2048

typedef __bf16 bf16_t;
typedef __bf16 v8bf  __attribute__((ext_vector_type(8)));
typedef __bf16 v16bf __attribute__((ext_vector_type(16)));
typedef float  v8f   __attribute__((ext_vector_type(8)));
typedef int    v4i   __attribute__((ext_vector_type(4)));

// ---- async global->LDS (CDNA5) with graceful fallback --------------------
#if defined(__has_builtin)
#  if __has_builtin(__builtin_amdgcn_global_load_async_to_lds_b128)
#    define HAVE_ASYNC_LDS 1
#  endif
#endif
#ifndef HAVE_ASYNC_LDS
#  define HAVE_ASYNC_LDS 0
#endif

typedef __attribute__((address_space(1))) v4i* as1_v4i;   // global (prints as __device__)
typedef __attribute__((address_space(3))) v4i* as3_v4i;   // LDS

__device__ __forceinline__ void stage16(const bf16_t* g, bf16_t* l) {
#if HAVE_ASYNC_LDS
  __builtin_amdgcn_global_load_async_to_lds_b128((as1_v4i)g, (as3_v4i)l, 0, 0);
#else
  *(v8bf*)l = *(const v8bf*)g;
#endif
}

__device__ __forceinline__ void wait_stage() {
#if HAVE_ASYNC_LDS
#  if __has_builtin(__builtin_amdgcn_s_wait_asynccnt)
  __builtin_amdgcn_s_wait_asynccnt(0);
#  else
  asm volatile("s_wait_asynccnt 0x0" ::: "memory");
#  endif
#endif
}

__device__ __forceinline__ v16bf cat16(v8bf lo, v8bf hi) {
  return __builtin_shufflevector(lo, hi, 0,1,2,3,4,5,6,7,8,9,10,11,12,13,14,15);
}

// ---------------------------------------------------------------------------
// fp32 -> bf16 conversion (activations, row-major kept)
// ---------------------------------------------------------------------------
__global__ void f32_to_bf16_kernel(const float* __restrict__ in,
                                   bf16_t* __restrict__ out, size_t n) {
  size_t i = (size_t)blockIdx.x * 256 + threadIdx.x;
  if (i < n) out[i] = (bf16_t)in[i];
}

// fp32 (L,K,N) -> bf16 (L,N,K): weights converted AND transposed once, so the
// GEMM's B-fragments are contiguous-K 16B LDS loads (same shape as A-fragments).
__global__ void f32_to_bf16_tr_kernel(const float* __restrict__ in,
                                      bf16_t* __restrict__ out,
                                      int K_, int N_, size_t total) {
  size_t i = (size_t)blockIdx.x * 256 + threadIdx.x;
  if (i >= total) return;
  int n = (int)(i % N_);
  size_t t = i / N_;
  int k = (int)(t % K_);
  int l = (int)(t / K_);
  out[((size_t)l * N_ + n) * K_ + k] = (bf16_t)in[i];
}

// ---------------------------------------------------------------------------
// bf16 WMMA GEMM:  C[M,N] = alpha * (A[M,K] @ B[K,N] + bias[N])
//   A:  [M][K] row-major bf16
//   Bt: [N][K] row-major bf16 (i.e. B transposed)
// Block = 128 threads (4 waves, 2x2), block tile 32(M) x 128(N), K-step 32.
// Each wave computes a 16x64 strip -> 4 accumulators -> 4 wmma per K-step.
// grid = (M/32, N/128); M%32==0, N%128==0, K%32==0.
// ---------------------------------------------------------------------------
__global__ void __launch_bounds__(128) gemm_bf16_wmma(
    const bf16_t* __restrict__ Ap, const bf16_t* __restrict__ Btp,
    const float* __restrict__ bias, float* __restrict__ Cp,
    int N, int K, float alpha)
{
  __shared__ __align__(16) bf16_t As[32][40];    // 32x32 A tile (pad keeps 16B align)
  __shared__ __align__(16) bf16_t Bs[128][40];   // 128(n) x 32(k) Bt tile

  const int tid  = threadIdx.x;
  const int wave = tid >> 5;
  const int lane = tid & 31;
  const int half = lane >> 4;
  const int idx  = lane & 15;
  const int wm   = wave >> 1;                    // 0..1 : which 16-row strip
  const int wn   = wave & 1;                     // 0..1 : which 64-col strip
  const int m0 = blockIdx.x * 32;
  const int n0 = blockIdx.y * 128;

  const int ar = tid >> 2, ac = (tid & 3) * 8;   // per-thread A chunk (16B)

  v8f acc0 = {}, acc1 = {}, acc2 = {}, acc3 = {};

  for (int k0 = 0; k0 < K; k0 += 32) {
    // Stage A tile: 32x32 bf16 = 128 x 16B chunks, one per thread
    stage16(&Ap[(size_t)(m0 + ar) * K + (k0 + ac)], &As[ar][ac]);
    // Stage Bt tile: 128x32 bf16 = 512 x 16B chunks, four per thread
    #pragma unroll
    for (int it = 0; it < 4; ++it) {
      int q = tid + 128 * it;
      int br = q >> 2, bc = (q & 3) * 8;
      stage16(&Btp[(size_t)(n0 + br) * K + (k0 + bc)], &Bs[br][bc]);
    }
    if (k0 + 32 < K) {   // prefetch next K tile -> global_prefetch_b8
      __builtin_prefetch(&Ap[(size_t)(m0 + ar) * K + (k0 + 32)], 0, 1);
      __builtin_prefetch(&Btp[(size_t)(n0 + tid) * K + (k0 + 32)], 0, 1);
    }
    wait_stage();
    __syncthreads();

    // A fragment (16x32): lanes 0-15 row=idx K{0..15}; lanes 16-31 K{16..31}
    const int arow = wm * 16 + idx;
    const int akb  = half * 16;
    v16bf a = cat16(*(const v8bf*)&As[arow][akb],
                    *(const v8bf*)&As[arow][akb + 8]);

    // B fragments: col = wn*64 + t*16 + idx, lane half selects K {0..15}/{16..31}
    const int nb = wn * 64 + idx;
    const int bk = half * 16;
    v16bf b0 = cat16(*(const v8bf*)&Bs[nb     ][bk], *(const v8bf*)&Bs[nb     ][bk + 8]);
    v16bf b1 = cat16(*(const v8bf*)&Bs[nb + 16][bk], *(const v8bf*)&Bs[nb + 16][bk + 8]);
    v16bf b2 = cat16(*(const v8bf*)&Bs[nb + 32][bk], *(const v8bf*)&Bs[nb + 32][bk + 8]);
    v16bf b3 = cat16(*(const v8bf*)&Bs[nb + 48][bk], *(const v8bf*)&Bs[nb + 48][bk + 8]);

    acc0 = __builtin_amdgcn_wmma_f32_16x16x32_bf16(false, a, false, b0, (short)0, acc0, false, false);
    acc1 = __builtin_amdgcn_wmma_f32_16x16x32_bf16(false, a, false, b1, (short)0, acc1, false, false);
    acc2 = __builtin_amdgcn_wmma_f32_16x16x32_bf16(false, a, false, b2, (short)0, acc2, false, false);
    acc3 = __builtin_amdgcn_wmma_f32_16x16x32_bf16(false, a, false, b3, (short)0, acc3, false, false);
    __syncthreads();
  }

  // C layout per tile: VGPR v -> row = v + half*8, col = lane&15
  const int rbase = m0 + wm * 16 + half * 8;
  const int cbase = n0 + wn * 64 + idx;
  #pragma unroll
  for (int t = 0; t < 4; ++t) {
    const v8f* at = (t == 0) ? &acc0 : (t == 1) ? &acc1 : (t == 2) ? &acc2 : &acc3;
    const int col = cbase + t * 16;
    const float bval = bias ? bias[col] : 0.0f;
    #pragma unroll
    for (int v = 0; v < 8; ++v)
      Cp[(size_t)(rbase + v) * N + col] = alpha * ((*at)[v] + bval);
  }
}

// ---------------------------------------------------------------------------
// scores[b,i,h,j] = sum_d ((q[b,i,h,d] + r[b,j,i,h*D+d]) * s) * (k[b,j,h,d] + r[b,j,i,E+h*D+d])
// q already carries the first *s from its projection (double-scaling is intentional).
// ---------------------------------------------------------------------------
__global__ void scores_kernel(const float* __restrict__ q, const float* __restrict__ k,
                              const float* __restrict__ r, float* __restrict__ sc, float s)
{
  int idx = blockIdx.x * 256 + threadIdx.x;       // over NB*NS*NS*NH = 262144
  int h = idx & (NH - 1);
  int j = (idx >> 3) & (NS - 1);
  int i = (idx >> 10) & (NS - 1);
  int b = idx >> 17;
  const float* qp = q + ((size_t)(b * NS + i)) * NE + h * ND;
  const float* kp = k + ((size_t)(b * NS + j)) * NE + h * ND;
  const float* rp = r + ((size_t)((b * NS + j) * NS + i)) * (2 * NE) + h * ND;
  float acc = 0.0f;
  #pragma unroll 8
  for (int d = 0; d < ND; ++d) {
    float qa = (qp[d] + rp[d]) * s;
    float ka = kp[d] + rp[NE + d];
    acc = fmaf(qa, ka, acc);
  }
  sc[((size_t)(b * NS + i) * NH + h) * NS + j] = acc;
}

// Softmax over j for each (b,i,h) row of length NS=128.
__global__ void softmax_kernel(float* __restrict__ sc) {
  __shared__ float red[NS];
  const int j = threadIdx.x;
  float* p = sc + (size_t)blockIdx.x * NS;
  float v = p[j];
  red[j] = v;
  __syncthreads();
  for (int o = NS / 2; o > 0; o >>= 1) {
    if (j < o) red[j] = fmaxf(red[j], red[j + o]);
    __syncthreads();
  }
  const float mx = red[0];
  __syncthreads();
  const float e = expf(v - mx);
  red[j] = e;
  __syncthreads();
  for (int o = NS / 2; o > 0; o >>= 1) {
    if (j < o) red[j] += red[j + o];
    __syncthreads();
  }
  p[j] = e / red[0];
}

// attn[b,i,h*D+d] = sum_j w[b,i,h,j] * v[b,j,h*D+d]
__global__ void attn_kernel(const float* __restrict__ w, const float* __restrict__ v,
                            float* __restrict__ out)
{
  int idx = blockIdx.x * 256 + threadIdx.x;       // over NB*NS*NH*ND = 131072
  int d = idx & (ND - 1);
  int h = (idx >> 6) & (NH - 1);
  int i = (idx >> 9) & (NS - 1);
  int b = idx >> 16;
  const float* wp = w + ((size_t)(b * NS + i) * NH + h) * NS;
  const float* vp = v + (size_t)(b * NS) * NE + h * ND + d;
  float acc = 0.0f;
  #pragma unroll 8
  for (int j = 0; j < NS; ++j) acc = fmaf(wp[j], vp[(size_t)j * NE], acc);
  out[(size_t)(b * NS + i) * NE + h * ND + d] = acc;
}

// x = LayerNorm(res + t) * g + be ; also emits bf16 copy for the next GEMM.
__global__ void add_ln_kernel(const float* __restrict__ res, const float* __restrict__ t,
                              const float* __restrict__ g, const float* __restrict__ be,
                              float* __restrict__ xout, bf16_t* __restrict__ xbf)
{
  __shared__ float red[256];
  const int row = blockIdx.x, tid = threadIdx.x;
  const float* rp = res + (size_t)row * NE;
  const float* tp = t + (size_t)row * NE;
  float v0 = rp[tid] + tp[tid];
  float v1 = rp[tid + 256] + tp[tid + 256];
  red[tid] = v0 + v1;
  __syncthreads();
  for (int o = 128; o > 0; o >>= 1) {
    if (tid < o) red[tid] += red[tid + o];
    __syncthreads();
  }
  const float mu = red[0] * (1.0f / NE);
  __syncthreads();
  const float d0 = v0 - mu, d1 = v1 - mu;
  red[tid] = d0 * d0 + d1 * d1;
  __syncthreads();
  for (int o = 128; o > 0; o >>= 1) {
    if (tid < o) red[tid] += red[tid + o];
    __syncthreads();
  }
  const float inv = rsqrtf(red[0] * (1.0f / NE) + 1e-5f);
  const float o0 = d0 * inv * g[tid] + be[tid];
  const float o1 = d1 * inv * g[tid + 256] + be[tid + 256];
  xout[(size_t)row * NE + tid] = o0;
  xout[(size_t)row * NE + tid + 256] = o1;
  xbf[(size_t)row * NE + tid] = (bf16_t)o0;
  xbf[(size_t)row * NE + tid + 256] = (bf16_t)o1;
}

// exact-erf GELU, emitted directly as bf16 for the W2 GEMM
__global__ void gelu_bf16_kernel(const float* __restrict__ in, bf16_t* __restrict__ out, int n) {
  int i = blockIdx.x * 256 + threadIdx.x;
  if (i < n) {
    float x = in[i];
    out[i] = (bf16_t)(x * 0.5f * (1.0f + erff(x * 0.7071067811865476f)));
  }
}

// ---------------------------------------------------------------------------
// Host orchestration
// ---------------------------------------------------------------------------
extern "C" void kernel_launch(void* const* d_in, const int* in_sizes, int n_in,
                              void* d_out, int out_size, void* d_ws, size_t ws_size,
                              hipStream_t stream)
{
  (void)in_sizes; (void)n_in; (void)out_size; (void)ws_size;

  const float* x_in = (const float*)d_in[0];
  const float* rel  = (const float*)d_in[1];
  const float* Wq = (const float*)d_in[2];  const float* bq = (const float*)d_in[3];
  const float* Wk = (const float*)d_in[4];  const float* bk = (const float*)d_in[5];
  const float* Wv = (const float*)d_in[6];  const float* bv = (const float*)d_in[7];
  const float* Wr = (const float*)d_in[8];
  const float* Wo = (const float*)d_in[9];  const float* bo = (const float*)d_in[10];
  const float* W1 = (const float*)d_in[11]; const float* b1 = (const float*)d_in[12];
  const float* W2 = (const float*)d_in[13]; const float* b2 = (const float*)d_in[14];
  const float* ln1g = (const float*)d_in[15]; const float* ln1b = (const float*)d_in[16];
  const float* ln2g = (const float*)d_in[17]; const float* ln2b = (const float*)d_in[18];

  char* base = (char*)d_ws;
  size_t off = 0;
  auto alloc = [&](size_t elems, size_t esz) -> void* {
    void* p = base + off;
    off += (elems * esz + 255) & ~(size_t)255;
    return p;
  };

  const size_t REL_N = (size_t)NB * NS * NS * NE;        // 16,777,216
  bf16_t* rel_bf = (bf16_t*)alloc(REL_N, 2);
  float*  r_f32  = (float*) alloc((size_t)NB * NS * NS * 2 * NE, 4);   // 134 MB
  // transposed bf16 weights: (L, N, K)
  bf16_t* wq_t  = (bf16_t*)alloc((size_t)NL * NE * NE, 2);
  bf16_t* wk_t  = (bf16_t*)alloc((size_t)NL * NE * NE, 2);
  bf16_t* wv_t  = (bf16_t*)alloc((size_t)NL * NE * NE, 2);
  bf16_t* wo_t  = (bf16_t*)alloc((size_t)NL * NE * NE, 2);
  bf16_t* wr_t  = (bf16_t*)alloc((size_t)NL * NE * 2 * NE, 2);
  bf16_t* w1_t  = (bf16_t*)alloc((size_t)NL * NE * NF, 2);
  bf16_t* w2_t  = (bf16_t*)alloc((size_t)NL * NF * NE, 2);
  float*  x_f    = (float*) alloc((size_t)NB * NS * NE, 4);
  bf16_t* x_bf   = (bf16_t*)alloc((size_t)NB * NS * NE, 2);
  float*  qf     = (float*) alloc((size_t)NB * NS * NE, 4);
  float*  kf     = (float*) alloc((size_t)NB * NS * NE, 4);
  float*  vf     = (float*) alloc((size_t)NB * NS * NE, 4);
  float*  sc     = (float*) alloc((size_t)NB * NS * NH * NS, 4);
  float*  attn_f = (float*) alloc((size_t)NB * NS * NE, 4);
  bf16_t* attn_b = (bf16_t*)alloc((size_t)NB * NS * NE, 2);
  float*  t0     = (float*) alloc((size_t)NB * NS * NE, 4);
  float*  h1     = (float*) alloc((size_t)NB * NS * NF, 4);
  bf16_t* g_bf   = (bf16_t*)alloc((size_t)NB * NS * NF, 2);
  float*  t1     = (float*) alloc((size_t)NB * NS * NE, 4);

  auto cvt = [&](const float* in, bf16_t* out, size_t n) {
    f32_to_bf16_kernel<<<dim3((unsigned)((n + 255) / 256)), 256, 0, stream>>>(in, out, n);
  };
  auto cvt_tr = [&](const float* in, bf16_t* out, int K_, int N_) {
    size_t n = (size_t)NL * K_ * N_;
    f32_to_bf16_tr_kernel<<<dim3((unsigned)((n + 255) / 256)), 256, 0, stream>>>(in, out, K_, N_, n);
  };

  // One-time conversions (weights transposed to (L,N,K); relation row-major)
  cvt(rel, rel_bf, REL_N);
  cvt_tr(Wq, wq_t, NE, NE);
  cvt_tr(Wk, wk_t, NE, NE);
  cvt_tr(Wv, wv_t, NE, NE);
  cvt_tr(Wo, wo_t, NE, NE);
  cvt_tr(Wr, wr_t, NE, 2 * NE);
  cvt_tr(W1, w1_t, NE, NF);
  cvt_tr(W2, w2_t, NF, NE);
  (void)hipMemcpyAsync(x_f, x_in, (size_t)NB * NS * NE * 4, hipMemcpyDeviceToDevice, stream);
  cvt(x_in, x_bf, (size_t)NB * NS * NE);

  const int M = NB * NS;          // 256
  const float s = 0.125f;         // D^-0.5, D = 64

  for (int l = 0; l < NL; ++l) {
    // Q/K/V projections (q carries the first *s, applied after bias as in ref)
    gemm_bf16_wmma<<<dim3(M / 32, NE / 128), 128, 0, stream>>>(
        x_bf, wq_t + (size_t)l * NE * NE, bq + (size_t)l * NE, qf, NE, NE, s);
    gemm_bf16_wmma<<<dim3(M / 32, NE / 128), 128, 0, stream>>>(
        x_bf, wk_t + (size_t)l * NE * NE, bk + (size_t)l * NE, kf, NE, NE, 1.0f);
    gemm_bf16_wmma<<<dim3(M / 32, NE / 128), 128, 0, stream>>>(
        x_bf, wv_t + (size_t)l * NE * NE, bv + (size_t)l * NE, vf, NE, NE, 1.0f);

    // r = relation @ Wr[l] : (32768 x 512) @ (512 x 1024) — the dominant GEMM
    gemm_bf16_wmma<<<dim3((NB * NS * NS) / 32, (2 * NE) / 128), 128, 0, stream>>>(
        rel_bf, wr_t + (size_t)l * NE * 2 * NE, nullptr, r_f32, 2 * NE, NE, 1.0f);

    // scores -> softmax(j) -> attention-weighted sum over v
    scores_kernel<<<dim3((NB * NS * NS * NH) / 256), 256, 0, stream>>>(qf, kf, r_f32, sc, s);
    softmax_kernel<<<dim3(NB * NS * NH), NS, 0, stream>>>(sc);
    attn_kernel<<<dim3((NB * NS * NH * ND) / 256), 256, 0, stream>>>(sc, vf, attn_f);

    // out projection + residual + LN1
    cvt(attn_f, attn_b, (size_t)NB * NS * NE);
    gemm_bf16_wmma<<<dim3(M / 32, NE / 128), 128, 0, stream>>>(
        attn_b, wo_t + (size_t)l * NE * NE, bo + (size_t)l * NE, t0, NE, NE, 1.0f);
    add_ln_kernel<<<dim3(M), 256, 0, stream>>>(
        x_f, t0, ln1g + (size_t)l * NE, ln1b + (size_t)l * NE, x_f, x_bf);

    // FFN: GELU(x@W1+b1)@W2+b2, residual + LN2
    gemm_bf16_wmma<<<dim3(M / 32, NF / 128), 128, 0, stream>>>(
        x_bf, w1_t + (size_t)l * NE * NF, b1 + (size_t)l * NF, h1, NF, NE, 1.0f);
    gelu_bf16_kernel<<<dim3((NB * NS * NF) / 256), 256, 0, stream>>>(h1, g_bf, NB * NS * NF);
    gemm_bf16_wmma<<<dim3(M / 32, NE / 128), 128, 0, stream>>>(
        g_bf, w2_t + (size_t)l * NF * NE, b2 + (size_t)l * NE, t1, NE, NF, 1.0f);
    float* xdst = (l == NL - 1) ? (float*)d_out : x_f;
    add_ln_kernel<<<dim3(M), 256, 0, stream>>>(
        x_f, t1, ln2g + (size_t)l * NE, ln2b + (size_t)l * NE, xdst, x_bf);
  }
}
